// NTXent_678604833492
// MI455X (gfx1250) — matrix operand order repaired and checked by reading, not compiled, not run on hardware
//
#include <hip/hip_runtime.h>
#include <math.h>

#define BATCH 4096
#define NTOT  8192          // 2*BATCH
#define DIM   128
#define CHUNKS 8
#define TILES_PER_CHUNK ((NTOT / 16) / CHUNKS)   // 64
#define INV_TEMP 2.0f       // 1 / 0.5
#define SHIFT    2.0f       // sim/TEMP is bounded by 2, fixed logsumexp shift

typedef __bf16 bf16;
typedef __attribute__((ext_vector_type(16))) __bf16 v16bf;
typedef __attribute__((ext_vector_type(4)))  __bf16 bf16x4;
typedef __attribute__((ext_vector_type(8)))  float  v8f;
typedef __attribute__((ext_vector_type(4)))  unsigned int u32x4;

union BfFrag { v16bf v; u32x4 q[2]; };

// ---------------------------------------------------------------------------
// Kernel 1: normalize rows of z = concat(zx, zy), split into bf16 hi + lo
// one wave per row, 4 features per lane
// ---------------------------------------------------------------------------
__global__ __launch_bounds__(256) void ntxent_prep(const float* __restrict__ zx,
                                                   const float* __restrict__ zy,
                                                   unsigned short* __restrict__ zh_u,
                                                   unsigned short* __restrict__ zl_u) {
    bf16* zh = reinterpret_cast<bf16*>(zh_u);
    bf16* zl = reinterpret_cast<bf16*>(zl_u);
    const int wave = threadIdx.x >> 5;
    const int lane = threadIdx.x & 31;
    const int row  = blockIdx.x * 8 + wave;           // 0..8191
    const float* src = (row < BATCH) ? (zx + (size_t)row * DIM)
                                     : (zy + (size_t)(row - BATCH) * DIM);
    float4 v = reinterpret_cast<const float4*>(src)[lane];   // features lane*4..+3
    float ss = v.x * v.x + v.y * v.y + v.z * v.z + v.w * v.w;
    #pragma unroll
    for (int off = 16; off >= 1; off >>= 1) ss += __shfl_xor(ss, off, 32);
    const float inv = 1.0f / fmaxf(sqrtf(ss), 1e-8f);
    float zn[4] = { v.x * inv, v.y * inv, v.z * inv, v.w * inv };
    bf16x4 h, l;
    #pragma unroll
    for (int k = 0; k < 4; ++k) {
        bf16 hi = (bf16)zn[k];
        h[k] = hi;
        l[k] = (bf16)(zn[k] - (float)hi);
    }
    *reinterpret_cast<bf16x4*>(zh + (size_t)row * DIM + lane * 4) = h;
    *reinterpret_cast<bf16x4*>(zl + (size_t)row * DIM + lane * 4) = l;
}

// ---------------------------------------------------------------------------
// Kernel 2: positives  pos[i] = (1/TEMP) * dot(zn_i, zn_{(i+BATCH)%N})
// one wave per row
// ---------------------------------------------------------------------------
__global__ __launch_bounds__(256) void ntxent_pos(const unsigned short* __restrict__ zh_u,
                                                  const unsigned short* __restrict__ zl_u,
                                                  float* __restrict__ pos) {
    const bf16* zh = reinterpret_cast<const bf16*>(zh_u);
    const bf16* zl = reinterpret_cast<const bf16*>(zl_u);
    const int wave = threadIdx.x >> 5;
    const int lane = threadIdx.x & 31;
    const int i = blockIdx.x * 8 + wave;              // 0..8191
    const int p = (i + BATCH) & (NTOT - 1);
    const int d = lane * 4;
    float dot = 0.f;
    #pragma unroll
    for (int k = 0; k < 4; ++k) {
        float a = (float)zh[(size_t)i * DIM + d + k] + (float)zl[(size_t)i * DIM + d + k];
        float b = (float)zh[(size_t)p * DIM + d + k] + (float)zl[(size_t)p * DIM + d + k];
        dot += a * b;
    }
    #pragma unroll
    for (int off = 16; off >= 1; off >>= 1) dot += __shfl_xor(dot, off, 32);
    if (lane == 0) pos[i] = INV_TEMP * dot;
}

// ---------------------------------------------------------------------------
// Kernel 3: fused GEMM (split-bf16 WMMA) + fixed-shift sum-exp over columns.
// block = one 32-row pair of strips; wave w handles column-chunk w (64 tiles).
// Each B fragment load feeds TWO 16x16 output tiles (M=32): 24 WMMA / 16 loads.
// sim tile: D = sum_k ( Ahi*Bhi + Ahi*Blo + Alo*Bhi ), fp32 accumulate.
// ---------------------------------------------------------------------------
__global__ __launch_bounds__(256) void ntxent_main(const unsigned short* __restrict__ zh_u,
                                                   const unsigned short* __restrict__ zl_u,
                                                   float* __restrict__ partS) {
    const bf16* zh = reinterpret_cast<const bf16*>(zh_u);
    const bf16* zl = reinterpret_cast<const bf16*>(zl_u);
    const int wave  = threadIdx.x >> 5;
    const int lane  = threadIdx.x & 31;
    const int pair  = blockIdx.x;                     // 0..255
    const int chunk = wave;                           // 0..7
    const int half  = lane >> 4;                      // 0 or 1
    const int n     = lane & 15;
    const int strip0 = pair * 2;                      // 16-row tile indices
    const int strip1 = pair * 2 + 1;

    // --- preload A fragments for both strips (constant over column loop) ---
    // ISA 16-bit A 16x32 layout: lane (l&15)=M; lanes0-15: K 0..7 (v0-3),
    // K 16..23 (v4-7); lanes16-31: K 8..15 and K 24..31.
    BfFrag Ah[2][4], Al[2][4];
    #pragma unroll
    for (int j = 0; j < 2; ++j) {
        const int arow = (pair * 2 + j) * 16 + n;
        const bf16* aph = zh + (size_t)arow * DIM;
        const bf16* apl = zl + (size_t)arow * DIM;
        #pragma unroll
        for (int kc = 0; kc < 4; ++kc) {
            const int fb = kc * 32 + half * 8;
            Ah[j][kc].q[0] = *reinterpret_cast<const u32x4*>(aph + fb);
            Ah[j][kc].q[1] = *reinterpret_cast<const u32x4*>(aph + fb + 16);
            Al[j][kc].q[0] = *reinterpret_cast<const u32x4*>(apl + fb);
            Al[j][kc].q[1] = *reinterpret_cast<const u32x4*>(apl + fb + 16);
        }
    }

    // diagonal lane mask (D layout: VGPR r, lane l -> row r+8*half, col n)
    float dmask[8];
    #pragma unroll
    for (int r = 0; r < 8; ++r) dmask[r] = ((r + half * 8) == n) ? 0.f : 1.f;

    // per-lane running sum of exp(v - SHIFT); cross-lane summed once at end
    float S0[8], S1[8];
    #pragma unroll
    for (int r = 0; r < 8; ++r) { S0[r] = 0.f; S1[r] = 0.f; }

    const int colTile0 = chunk * TILES_PER_CHUNK;
    for (int t = 0; t < TILES_PER_CHUNK; ++t) {
        const int tile = colTile0 + t;
        const int colBase = tile * 16;
        // B 32x16 layout: lane (l&15)=N -> row colBase+N; lanes0-15 hold
        // K 0..7 and K 16..23, lanes16-31 hold K 8..15 and K 24..31 (same as A).
        const bf16* bph = zh + (size_t)(colBase + n) * DIM;
        const bf16* bpl = zl + (size_t)(colBase + n) * DIM;

        v8f acc0 = {0.f, 0.f, 0.f, 0.f, 0.f, 0.f, 0.f, 0.f};
        v8f acc1 = {0.f, 0.f, 0.f, 0.f, 0.f, 0.f, 0.f, 0.f};
        #pragma unroll
        for (int kc = 0; kc < 4; ++kc) {
            const int fb = kc * 32 + half * 8;
            BfFrag Bh, Bl;
            Bh.q[0] = *reinterpret_cast<const u32x4*>(bph + fb);
            Bh.q[1] = *reinterpret_cast<const u32x4*>(bph + fb + 16);
            Bl.q[0] = *reinterpret_cast<const u32x4*>(bpl + fb);
            Bl.q[1] = *reinterpret_cast<const u32x4*>(bpl + fb + 16);
            acc0 = __builtin_amdgcn_wmma_f32_16x16x32_bf16(false, Ah[0][kc].v, false, Bh.v,
                                                           (short)0, acc0, false, false);
            acc1 = __builtin_amdgcn_wmma_f32_16x16x32_bf16(false, Ah[1][kc].v, false, Bh.v,
                                                           (short)0, acc1, false, false);
            acc0 = __builtin_amdgcn_wmma_f32_16x16x32_bf16(false, Ah[0][kc].v, false, Bl.v,
                                                           (short)0, acc0, false, false);
            acc1 = __builtin_amdgcn_wmma_f32_16x16x32_bf16(false, Ah[1][kc].v, false, Bl.v,
                                                           (short)0, acc1, false, false);
            acc0 = __builtin_amdgcn_wmma_f32_16x16x32_bf16(false, Al[0][kc].v, false, Bh.v,
                                                           (short)0, acc0, false, false);
            acc1 = __builtin_amdgcn_wmma_f32_16x16x32_bf16(false, Al[1][kc].v, false, Bh.v,
                                                           (short)0, acc1, false, false);
        }

        const bool d0 = (tile == strip0);             // wave-uniform diagonal hit
        const bool d1 = (tile == strip1);
        #pragma unroll
        for (int r = 0; r < 8; ++r) {
            float e0 = __expf(acc0[r] * INV_TEMP - SHIFT);   // bounded: <= 1
            float e1 = __expf(acc1[r] * INV_TEMP - SHIFT);
            if (d0) e0 *= dmask[r];                   // exclude self (diagonal)
            if (d1) e1 *= dmask[r];
            S0[r] += e0;
            S1[r] += e1;
        }
    }

    // one cross-lane (16-lane half) sum per row, then write partials
    #pragma unroll
    for (int r = 0; r < 8; ++r) {
        #pragma unroll
        for (int off = 8; off >= 1; off >>= 1) {
            S0[r] += __shfl_xor(S0[r], off, 32);
            S1[r] += __shfl_xor(S1[r], off, 32);
        }
    }
    if (n == 0) {                                     // lanes 0 and 16 write
        #pragma unroll
        for (int r = 0; r < 8; ++r) {
            const int m = r + half * 8;
            partS[(size_t)(strip0 * 16 + m) * CHUNKS + chunk] = S0[r];
            partS[(size_t)(strip1 * 16 + m) * CHUNKS + chunk] = S1[r];
        }
    }
}

// ---------------------------------------------------------------------------
// Kernel 4: combine chunk partials, subtract positives, deterministic
// single-block reduction to the scalar loss.  LSE_i = SHIFT + log(sum S).
// ---------------------------------------------------------------------------
__global__ __launch_bounds__(256) void ntxent_reduce(const float* __restrict__ partS,
                                                     const float* __restrict__ pos,
                                                     float* __restrict__ out) {
    float acc = 0.f;
    for (int k = 0; k < NTOT / 256; ++k) {
        const int i = threadIdx.x + k * 256;
        float S = 0.f;
        #pragma unroll
        for (int c = 0; c < CHUNKS; ++c) S += partS[(size_t)i * CHUNKS + c];
        acc += (SHIFT + logf(S)) - pos[i];
    }
    #pragma unroll
    for (int off = 16; off >= 1; off >>= 1) acc += __shfl_xor(acc, off, 32);
    __shared__ float red[8];
    if ((threadIdx.x & 31) == 0) red[threadIdx.x >> 5] = acc;
    __syncthreads();
    if (threadIdx.x == 0) {
        float t = 0.f;
        #pragma unroll
        for (int w = 0; w < 8; ++w) t += red[w];
        out[0] = t * (1.0f / (float)NTOT);
    }
}

// ---------------------------------------------------------------------------
extern "C" void kernel_launch(void* const* d_in, const int* in_sizes, int n_in,
                              void* d_out, int out_size, void* d_ws, size_t ws_size,
                              hipStream_t stream) {
    (void)in_sizes; (void)n_in; (void)out_size; (void)ws_size;
    const float* zx = (const float*)d_in[0];
    const float* zy = (const float*)d_in[1];
    float* out = (float*)d_out;

    char* ws = (char*)d_ws;
    const size_t zbytes = (size_t)NTOT * DIM * sizeof(unsigned short);  // 2 MB each
    unsigned short* zh = (unsigned short*)(ws);
    unsigned short* zl = (unsigned short*)(ws + zbytes);
    float* pos   = (float*)(ws + 2 * zbytes);                           // 32 KB
    float* partS = (float*)(ws + 2 * zbytes + (size_t)NTOT * 4);        // 256 KB

    ntxent_prep  <<<NTOT / 8, 256, 0, stream>>>(zx, zy, zh, zl);
    ntxent_pos   <<<NTOT / 8, 256, 0, stream>>>(zh, zl, pos);
    ntxent_main  <<<NTOT / 32, 256, 0, stream>>>(zh, zl, partS);
    ntxent_reduce<<<1, 256, 0, stream>>>(partS, pos, out);
}
